// MissingAwareHeteroClassifier_11398843203884
// MI455X (gfx1250) — matrix operand order repaired and checked by reading, not compiled, not run on hardware
//
#include <hip/hip_runtime.h>
#include <hip/hip_bf16.h>

typedef __attribute__((ext_vector_type(16))) _Float16 v16h;
typedef __attribute__((ext_vector_type(8)))  float    v8f;

#define HID 128
#define NUM_GRAPHS 64

// ---------------------------------------------------------------------------
// zero fill
__global__ void zero_f32(float* __restrict__ p, long n) {
    long i = (long)blockIdx.x * blockDim.x + threadIdx.x;
    if (i < n) p[i] = 0.0f;
}

// ---------------------------------------------------------------------------
// projection: h = relu(x @ W + b + x[:,1:2]*miss)
__global__ void proj_kernel(const float* __restrict__ x, const float* __restrict__ W,
                            const float* __restrict__ b, const float* __restrict__ miss,
                            float* __restrict__ h, int N) {
    int idx = blockIdx.x * blockDim.x + threadIdx.x;   // N*32 threads, 4 cols each
    int n  = idx >> 5;
    int c4 = (idx & 31) * 4;
    if (n >= N) return;
    float x0 = x[n * 2 + 0], x1 = x[n * 2 + 1];
#pragma unroll
    for (int j = 0; j < 4; j++) {
        int c = c4 + j;
        float v = x0 * W[c] + x1 * W[HID + c] + b[c] + x1 * miss[c];
        h[(long)n * HID + c] = fmaxf(v, 0.0f);
    }
}

// ---------------------------------------------------------------------------
// edge scatter: msg[dst] += h[src]; cnt[dst] += 1. One wave per edge (strided).
__global__ void edge_scatter(const float* __restrict__ h, const int* __restrict__ ei, int E,
                             float* __restrict__ msg, float* __restrict__ cnt) {
    int warp   = (blockIdx.x * blockDim.x + threadIdx.x) >> 5;
    int lane   = threadIdx.x & 31;
    int nwarps = (gridDim.x * blockDim.x) >> 5;
    for (int e = warp; e < E; e += nwarps) {
        int src = ei[e];
        int dst = ei[E + e];
        const float4 v = *(const float4*)(h + (long)src * HID + lane * 4);
        float* m = msg + (long)dst * HID + lane * 4;
        atomicAdd(m + 0, v.x);
        atomicAdd(m + 1, v.y);
        atomicAdd(m + 2, v.z);
        atomicAdd(m + 3, v.w);
        if (lane == 0) atomicAdd(cnt + dst, 1.0f);
    }
}

// ---------------------------------------------------------------------------
// Pack a 128x128 f32 weight (row-major, K x N) into the CDNA5 16-bit B-matrix
// fragment layout: for kchunk c (K/32), column n, sel (lane>=16), 16 halves
// K = c*32 + sel*8 + (t&7) + (t>>3)*16 stored contiguously -> one 32B load per lane.
__global__ void pack_w(const float* __restrict__ W, _Float16* __restrict__ out) {
    int idx = blockIdx.x * blockDim.x + threadIdx.x;   // 16384
    if (idx >= HID * HID) return;
    int t   = idx & 15;
    int sel = (idx >> 4) & 1;
    int n   = (idx >> 5) & 127;
    int c   = idx >> 12;
    out[idx] = (_Float16)W[(c * 32 + sel * 8 + (t & 7) + (t >> 3) * 16) * HID + n];
}

// ---------------------------------------------------------------------------
// Fused SAGE GEMM: out[tile 16 x 128] = relu( sum_s A_s @ B_s + bias0 (+bias1) )
// A_s : f32 [nrows x 128], optionally divided per-row by max(cnt,1)
// B_s : pre-packed f16 fragments (pack_w layout), K = 128
// One wave (32 threads) per 16-row tile; 8 v8f accumulators = 16x128 f32 out.
// NSEG is a compile-time constant so every segment's pointers stay direct
// kernel args -> address-space inference promotes all loads to global_load.
template <int NSEG>
__global__ __launch_bounds__(32) void sage_gemm_wmma(
    const float* __restrict__ A0, const float* __restrict__ C0, const _Float16* __restrict__ B0,
    const float* __restrict__ A1, const _Float16* __restrict__ B1,
    const float* __restrict__ A2, const float* __restrict__ C2, const _Float16* __restrict__ B2,
    const float* __restrict__ A3, const _Float16* __restrict__ B3,
    const float* __restrict__ bias0, const float* __restrict__ bias1,
    float* __restrict__ out, int nrows) {
    const int tile = blockIdx.x;
    const int lane = threadIdx.x & 31;
    const int row0 = tile * 16;
    const int mrow = lane & 15;
    const int sel  = lane >> 4;            // 0: K 0-7/16-23, 1: K 8-15/24-31
    const int row  = row0 + mrow;
    const int rowc = (row < nrows) ? row : (nrows - 1);

    v8f acc[8] = {};

    // process one segment (inlined; cnt==nullptr folds away after inlining)
    auto do_seg = [&](const float* __restrict__ A, const float* __restrict__ cnt,
                      const _Float16* __restrict__ B) {
        float scale = 1.0f;
        if (cnt) scale = 1.0f / fmaxf(cnt[rowc], 1.0f);
        const float* arow = A + (long)rowc * HID;
#pragma unroll
        for (int c = 0; c < 4; c++) {      // K chunks of 32
            const int klo = c * 32 + sel * 8;
            v16h afrag;
#pragma unroll
            for (int j = 0; j < 8; j++) afrag[j]     = (_Float16)(arow[klo + j] * scale);
#pragma unroll
            for (int j = 0; j < 8; j++) afrag[8 + j] = (_Float16)(arow[klo + 16 + j] * scale);
            // preload all 8 B fragments for this K chunk, then the WMMA chain
            v16h bf[8];
#pragma unroll
            for (int nt = 0; nt < 8; nt++)
                bf[nt] = *(const v16h*)(B + (((c * 128) + (nt * 16 + mrow)) * 2 + sel) * 16);
#pragma unroll
            for (int nt = 0; nt < 8; nt++)
                acc[nt] = __builtin_amdgcn_wmma_f32_16x16x32_f16(
                    false, afrag, false, bf[nt], (short)0, acc[nt], false, false);
        }
    };

    do_seg(A0, C0, B0);
    do_seg(A1, nullptr, B1);
    if (NSEG > 2) {
        do_seg(A2, C2, B2);
        do_seg(A3, nullptr, B3);
    }

    // epilogue: C/D layout — VGPR i: lanes 0-15 -> M=i, lanes 16-31 -> M=8+i
#pragma unroll
    for (int nt = 0; nt < 8; nt++) {
        const int col = nt * 16 + mrow;
        float b = bias0[col];
        if (NSEG > 2) b += bias1[col];
#pragma unroll
        for (int i = 0; i < 8; i++) {
            const int r = row0 + (sel ? (8 + i) : i);
            float v = fmaxf(acc[nt][i] + b, 0.0f);
            if (r < nrows) out[(long)r * HID + col] = v;
        }
    }
}

// ---------------------------------------------------------------------------
// graph pooling: pooled[batch[n], col_off + c] += h[n, c]
__global__ void pool_scatter(const float* __restrict__ h, const int* __restrict__ batch,
                             float* __restrict__ pooled, int col_off, int N) {
    int idx = blockIdx.x * blockDim.x + threadIdx.x;
    int n  = idx >> 5;
    int c4 = (idx & 31) * 4;
    if (n >= N) return;
    int g = batch[n];
    const float* hp = h + (long)n * HID + c4;
    float* pp = pooled + (long)g * 256 + col_off + c4;
    atomicAdd(pp + 0, hp[0]);
    atomicAdd(pp + 1, hp[1]);
    atomicAdd(pp + 2, hp[2]);
    atomicAdd(pp + 3, hp[3]);
}

// ---------------------------------------------------------------------------
// head
__global__ void head1_kernel(const float* __restrict__ pooled, const float* __restrict__ W1,
                             const float* __restrict__ b1, float* __restrict__ z) {
    int g = blockIdx.x;      // 64
    int c = threadIdx.x;     // 128
    float acc = b1[c];
    for (int k = 0; k < 256; k++) acc += pooled[g * 256 + k] * W1[k * HID + c];
    z[g * HID + c] = fmaxf(acc, 0.0f);
}

__global__ void head2_kernel(const float* __restrict__ z, const float* __restrict__ W2,
                             const float* __restrict__ b2, float* __restrict__ out) {
    int g = blockIdx.x;      // 64
    int o = threadIdx.x;     // 8
    float acc = b2[o];
    for (int k = 0; k < HID; k++) acc += z[g * HID + k] * W2[k * 8 + o];
    out[g * 8 + o] = acc;
}

// ---------------------------------------------------------------------------
extern "C" void kernel_launch(void* const* d_in, const int* in_sizes, int n_in,
                              void* d_out, int out_size, void* d_ws, size_t ws_size,
                              hipStream_t stream) {
    const int N = in_sizes[0] / 2;     // 100000
    const int E = in_sizes[30] / 2;    // 1600000

    auto F = [&](int i) { return (const float*)d_in[i]; };
    auto I = [&](int i) { return (const int*)d_in[i]; };

    // input index map (setup_inputs dict order, params in insertion order)
    // 0 x_gene, 1 x_protein,
    // 2 proj_W_gene, 3 proj_b_gene, 4 miss_gene, 5 proj_W_protein, 6 proj_b_protein, 7 miss_protein,
    // 8.. L{i}_{et}_{Wl,bl,Wr}: idx = 8 + 9*i + 3*e (+0 Wl, +1 bl, +2 Wr), e: gg=0 gp=1 pp=2
    // 26 head_W1, 27 head_b1, 28 head_W2, 29 head_b2,
    // 30 ei_gg, 31 ei_gp, 32 ei_pp, 33 batch_gene, 34 batch_protein

    // workspace layout
    char*  ws  = (char*)d_ws;
    size_t off = 0;
    auto alloc = [&](size_t bytes) -> char* {
        char* p = ws + off;
        off += (bytes + 255) & ~(size_t)255;
        return p;
    };
    const size_t hbytes = (size_t)N * HID * sizeof(float);
    float*    h_g    = (float*)alloc(hbytes);
    float*    h_p    = (float*)alloc(hbytes);
    float*    h_g2   = (float*)alloc(hbytes);
    float*    h_p2   = (float*)alloc(hbytes);
    float*    msg0   = (float*)alloc(hbytes);
    float*    msg1   = (float*)alloc(hbytes);
    float*    cnt0   = (float*)alloc((size_t)N * sizeof(float));
    float*    cnt1   = (float*)alloc((size_t)N * sizeof(float));
    _Float16* wpk    = (_Float16*)alloc((size_t)12 * HID * HID * sizeof(_Float16));
    float*    pooled = (float*)alloc((size_t)NUM_GRAPHS * 256 * sizeof(float));
    float*    zbuf   = (float*)alloc((size_t)NUM_GRAPHS * HID * sizeof(float));

    auto wslot = [&](int layer, int e, int lr) {   // lr: 0=Wl, 1=Wr
        return wpk + (size_t)((layer * 3 + e) * 2 + lr) * HID * HID;
    };

    // pack all 12 layer weight matrices into WMMA B-fragment layout (f16)
    for (int i = 0; i < 2; i++)
        for (int e = 0; e < 3; e++) {
            int base = 8 + 9 * i + 3 * e;
            pack_w<<<(HID * HID + 255) / 256, 256, 0, stream>>>(F(base + 0), wslot(i, e, 0));
            pack_w<<<(HID * HID + 255) / 256, 256, 0, stream>>>(F(base + 2), wslot(i, e, 1));
        }

    // projections
    {
        int threads = N * 32;
        int grid = (threads + 255) / 256;
        proj_kernel<<<grid, 256, 0, stream>>>(F(0), F(2), F(3), F(4), h_g, N);
        proj_kernel<<<grid, 256, 0, stream>>>(F(1), F(5), F(6), F(7), h_p, N);
    }

    const long nh = (long)N * HID;
    const int  zgrid_h = (int)((nh + 255) / 256);
    const int  zgrid_n = (N + 255) / 256;
    const int  sgrid   = 4096;              // edge-scatter blocks (grid-stride)
    const int  ggrid   = (N + 15) / 16;     // one wave per 16-row tile

    for (int i = 0; i < 2; i++) {
        const float* hg_in  = (i == 0) ? h_g  : h_g2;
        const float* hp_in  = (i == 0) ? h_p  : h_p2;
        float*       hg_out = (i == 0) ? h_g2 : h_g;
        float*       hp_out = (i == 0) ? h_p2 : h_p;
        int base = 8 + 9 * i;
        const float* bl_gg = F(base + 0 + 1);
        const float* bl_gp = F(base + 3 + 1);
        const float* bl_pp = F(base + 6 + 1);

        // gene output: sage(gg)
        zero_f32<<<zgrid_h, 256, 0, stream>>>(msg0, nh);
        zero_f32<<<zgrid_n, 256, 0, stream>>>(cnt0, N);
        edge_scatter<<<sgrid, 256, 0, stream>>>(hg_in, I(30), E, msg0, cnt0);
        sage_gemm_wmma<2><<<ggrid, 32, 0, stream>>>(
            msg0, cnt0, wslot(i, 0, 0),
            hg_in, wslot(i, 0, 1),
            nullptr, nullptr, nullptr,
            nullptr, nullptr,
            bl_gg, nullptr, hg_out, N);

        // protein output: sage(gp) + sage(pp), fused as one K=512 GEMM
        zero_f32<<<zgrid_h, 256, 0, stream>>>(msg0, nh);
        zero_f32<<<zgrid_n, 256, 0, stream>>>(cnt0, N);
        edge_scatter<<<sgrid, 256, 0, stream>>>(hg_in, I(31), E, msg0, cnt0);
        zero_f32<<<zgrid_h, 256, 0, stream>>>(msg1, nh);
        zero_f32<<<zgrid_n, 256, 0, stream>>>(cnt1, N);
        edge_scatter<<<sgrid, 256, 0, stream>>>(hp_in, I(32), E, msg1, cnt1);
        sage_gemm_wmma<4><<<ggrid, 32, 0, stream>>>(
            msg0, cnt0, wslot(i, 1, 0),
            hp_in, wslot(i, 1, 1),
            msg1, cnt1, wslot(i, 2, 0),
            hp_in, wslot(i, 2, 1),
            bl_gp, bl_pp, hp_out, N);
    }

    // pooling (layer-1 outputs live in h_g / h_p)
    zero_f32<<<(NUM_GRAPHS * 256 + 255) / 256, 256, 0, stream>>>(pooled, NUM_GRAPHS * 256);
    {
        int threads = N * 32;
        int grid = (threads + 255) / 256;
        pool_scatter<<<grid, 256, 0, stream>>>(h_g, I(33), pooled, 0, N);
        pool_scatter<<<grid, 256, 0, stream>>>(h_p, I(34), pooled, HID, N);
    }

    // head
    head1_kernel<<<NUM_GRAPHS, HID, 0, stream>>>(pooled, F(26), F(27), zbuf);
    head2_kernel<<<NUM_GRAPHS, 8, 0, stream>>>(zbuf, F(28), F(29), (float*)d_out);
}
